// ElasticBertSelfAttention_91225105367509
// MI455X (gfx1250) — matrix operand order repaired and checked by reading, not compiled, not run on hardware
//
#include <hip/hip_runtime.h>
#include <hip/hip_bf16.h>

// ---------------------------------------------------------------------------
// ElasticBert self-attention for MI455X (gfx1250, wave32, WMMA bf16,
// async-to-LDS double buffering).  B=8, S=1024, H=1024, NH=16, K=512, HD=64.
// ---------------------------------------------------------------------------

#define B_    8
#define S_    1024
#define H_    1024
#define NH_   16
#define KSEL  512
#define HD_   64

typedef __bf16 bf16_t;
typedef __attribute__((ext_vector_type(16))) __bf16 v16bf;
typedef __attribute__((ext_vector_type(8)))  __bf16 bf16x8;
typedef __attribute__((ext_vector_type(4)))  __bf16 bf16x4;
typedef __attribute__((ext_vector_type(8)))  float  v8f;

static __device__ inline v8f zero8() {
    v8f z;
#pragma unroll
    for (int i = 0; i < 8; ++i) z[i] = 0.0f;
    return z;
}

// Load a 16x32 bf16 A/B fragment from two 16-byte LDS runs.
static __device__ inline v16bf ld_frag(const bf16_t* p0, const bf16_t* p1) {
    bf16x8 a = *(const bf16x8*)p0;
    bf16x8 b = *(const bf16x8*)p1;
    v16bf r;
#pragma unroll
    for (int i = 0; i < 8; ++i) { r[i] = a[i]; r[i + 8] = b[i]; }
    return r;
}

// Exchange with partner lane (lane ^ 16) via ds_swizzle SWAPX16 (0x401F).
static __device__ inline float swap16(float x) {
    return __int_as_float(__builtin_amdgcn_ds_swizzle(__float_as_int(x), 0x401f));
}

// CDNA5 async copy: per-lane 16B global -> LDS, tracked by ASYNCcnt.
static __device__ inline void async_ld_b128(unsigned lds_byte_off, const void* g) {
    asm volatile("global_load_async_to_lds_b128 %0, %1, off"
                 :: "v"(lds_byte_off), "v"(g) : "memory");
}
#define S_WAIT_ASYNC_0 asm volatile("s_wait_asynccnt 0x0" ::: "memory")
#define S_WAIT_ASYNC_3 asm volatile("s_wait_asynccnt 0x3" ::: "memory")
#define S_WAIT_ASYNC_4 asm volatile("s_wait_asynccnt 0x4" ::: "memory")

static __device__ inline unsigned lds_off(const void* p) {
    return (unsigned)(uintptr_t)p;   // low 32 bits of flat shared addr = LDS offset
}

// ---------------------------------------------------------------------------
// Kernel 0: f32 -> bf16 conversion (hidden + weights), 4 elems/thread.
// ---------------------------------------------------------------------------
__global__ void cvt_bf16_kernel(const float* __restrict__ in,
                                bf16_t* __restrict__ out, int n) {
    int i = (blockIdx.x * 256 + threadIdx.x) * 4;
    if (i >= n) return;
    float4 x = *(const float4*)(in + i);
    bf16x4 y;
    y[0] = (bf16_t)x.x; y[1] = (bf16_t)x.y; y[2] = (bf16_t)x.z; y[3] = (bf16_t)x.w;
    *(bf16x4*)(out + i) = y;
}

// ---------------------------------------------------------------------------
// Kernel 1: stable selection of K tokens per batch.
// ---------------------------------------------------------------------------
__global__ void build_idx_kernel(const int* __restrict__ remain,
                                 int* __restrict__ gidx,
                                 float* __restrict__ gmul) {
    int b = blockIdx.x;
    if (threadIdx.x != 0) return;
    const int* rl = remain + b * S_;
    int*   gi = gidx + b * KSEL;
    float* gm = gmul + b * KSEL;
    int pos = 0;
    for (int t = 0; t < S_ && pos < KSEL; ++t)
        if (rl[t]) { gi[pos] = t; gm[pos] = 1.0f; ++pos; }
    for (int t = 0; t < S_ && pos < KSEL; ++t)
        if (!rl[t]) { gi[pos] = t; gm[pos] = 0.0f; ++pos; }
}

// ---------------------------------------------------------------------------
// Kernel 2: out[m][n] = (sum_k X[src(m)][k]*W[n][k]) * mul(m) + bias[n]
// bf16 in / bf16 out, f32 accumulate.  Block tile 128(M) x 64(N), 8 waves.
// Double-buffered async-to-LDS staging of X/W k-tiles (k-step 32).
// transpose_out: V layout (b,h,d,s) with contiguous 16B stores.
// ---------------------------------------------------------------------------
__global__ __launch_bounds__(256)
void gemm_bias_kernel(const bf16_t* __restrict__ X,
                      const bf16_t* __restrict__ W,
                      const float* __restrict__ bias,
                      bf16_t* __restrict__ out,
                      const int* __restrict__ gidx,
                      const float* __restrict__ gmul,
                      int rows_per_batch,
                      int transpose_out) {
    __shared__ alignas(16) bf16_t XL[2 * 128 * 32];   // 16 KB
    __shared__ alignas(16) bf16_t WL[2 * 64 * 32];    //  8 KB
    __shared__ int    rsrc[128];
    __shared__ float  rmulL[128];

    const int tid  = threadIdx.x;
    const int wave = tid >> 5;
    const int lane = tid & 31;
    const int hl   = lane >> 4;
    const int col  = lane & 15;

    const int M0 = blockIdx.y * 128;
    const int N0 = blockIdx.x * 64;

    if (tid < 128) {
        int r = M0 + tid;
        int src; float vm = 1.0f;
        if (gidx) {
            int b  = r / rows_per_batch;
            int kk = r % rows_per_batch;
            src = b * S_ + gidx[b * rows_per_batch + kk];
            vm  = gmul[b * rows_per_batch + kk];
        } else {
            src = r;
        }
        rsrc[tid]  = src;
        rmulL[tid] = vm;
    }
    __syncthreads();

    const unsigned xoff = lds_off(XL);
    const unsigned woff = lds_off(WL);

    // Issue one k-tile (X: 512 b128 ops, W: 256) -> 3 async instr per wave.
    auto issue_tile = [&](int buf, int kk) {
#pragma unroll
        for (int rep = 0; rep < 2; ++rep) {
            int o = tid + rep * 256;
            int r = o >> 2, c = (o & 3) * 8;
            async_ld_b128(xoff + buf * 8192 + o * 16,
                          X + (size_t)rsrc[r] * H_ + kk + c);
        }
        {
            int o = tid;
            int r = o >> 2, c = (o & 3) * 8;
            async_ld_b128(woff + buf * 4096 + o * 16,
                          W + (size_t)(N0 + r) * H_ + kk + c);
        }
    };

    v8f acc[4];
#pragma unroll
    for (int nt = 0; nt < 4; ++nt) acc[nt] = zero8();

    issue_tile(0, 0);
    const int NIT = H_ / 32;
    for (int it = 0; it < NIT; ++it) {
        const int buf = it & 1;
        if (it + 1 < NIT) { issue_tile(buf ^ 1, (it + 1) * 32); S_WAIT_ASYNC_3; }
        else              { S_WAIT_ASYNC_0; }
        __syncthreads();

        const bf16_t* XLb = XL + buf * 4096;
        const bf16_t* WLb = WL + buf * 2048;
        // A: lane row m = lane&15 of this wave's strip; k runs {hl*8, 16+hl*8}
        const bf16_t* ap = XLb + (wave * 16 + col) * 32 + hl * 8;
        v16bf A = ld_frag(ap, ap + 16);
#pragma unroll
        for (int nt = 0; nt < 4; ++nt) {
            const bf16_t* bp = WLb + (nt * 16 + col) * 32 + hl * 8;
            v16bf Bf = ld_frag(bp, bp + 16);
            acc[nt] = __builtin_amdgcn_wmma_f32_16x16x32_bf16(
                false, A, false, Bf, (short)0, acc[nt], false, false);
        }
        __syncthreads();   // protect buffer reuse by next iteration's issue
    }

    // Epilogue: validity multiplier + bias, bf16 store.
    if (transpose_out) {
        // n -> (h, d); rows -> (b, s); store (b,h,d,s): j-contiguous in s.
#pragma unroll
        for (int nt = 0; nt < 4; ++nt) {
            int n = N0 + nt * 16 + col;
            int hh = n >> 6, d = n & 63;
            float bv = bias[n];
            int row0 = M0 + wave * 16 + 8 * hl;   // 8-aligned
            int b    = row0 / S_, s0 = row0 % S_;
            bf16x8 pack;
#pragma unroll
            for (int j = 0; j < 8; ++j)
                pack[j] = (bf16_t)(acc[nt][j] * rmulL[wave * 16 + 8 * hl + j] + bv);
            *(bf16x8*)(out + ((size_t)(b * NH_ + hh) * HD_ + d) * S_ + s0) = pack;
        }
    } else {
#pragma unroll
        for (int nt = 0; nt < 4; ++nt) {
            int n = N0 + nt * 16 + col;
            float bv = bias[n];
#pragma unroll
            for (int j = 0; j < 8; ++j) {
                int row = M0 + wave * 16 + 8 * hl + j;
                out[(size_t)row * H_ + n] =
                    (bf16_t)(acc[nt][j] * rmulL[wave * 16 + 8 * hl + j] + bv);
            }
        }
    }
}

// ---------------------------------------------------------------------------
// Kernel 3: flash attention per (b, h, 128-q tile); 8 waves x 16 q rows.
// 64-key chunks, double-buffered async-to-LDS K/V staging.
// scores^T = K x Q^T (lane owns one q column -> per-lane softmax + one
// ds_swizzle SWAPX16); ctx^T = V^T x P^T with V pre-transposed in memory.
// ---------------------------------------------------------------------------
__global__ __launch_bounds__(256)
void attn_kernel(const bf16_t* __restrict__ Qb,     // (b, token, H) bf16
                 const bf16_t* __restrict__ Kb,     // (b, token, H) bf16
                 const bf16_t* __restrict__ Vt,     // (b, h, d, s)  bf16
                 const float*  __restrict__ mask,   // (b, s)
                 float* __restrict__ out) {         // (b, q, H) f32
    __shared__ alignas(16) bf16_t QL[128 * 64];      // 16 KB
    __shared__ alignas(16) bf16_t KL[2 * 64 * 64];   // 16 KB
    __shared__ alignas(16) bf16_t VT[2 * 64 * 64];   // 16 KB  (d-major, key contig)
    __shared__ float ML[2][64];

    const int tid  = threadIdx.x;
    const int wave = tid >> 5;
    const int lane = tid & 31;
    const int hl   = lane >> 4;
    const int col  = lane & 15;

    const int b  = blockIdx.z;
    const int h  = blockIdx.y;
    const int qbase = blockIdx.x * 128;
    const size_t vbase = (size_t)(b * NH_ + h) * HD_ * S_;

    const unsigned qoff = lds_off(QL);
    const unsigned koff = lds_off(KL);
    const unsigned voff = lds_off(VT);

    // Issue one 64-key chunk (K: 2 ops/wave, V: 2 ops/wave) + mask slice.
    auto issue_chunk = [&](int buf, int s0) {
#pragma unroll
        for (int rep = 0; rep < 2; ++rep) {
            int o = tid + rep * 256;
            int r = o >> 3, c = (o & 7) * 8;
            async_ld_b128(koff + buf * 8192 + o * 16,
                          Kb + ((size_t)(b * S_ + s0 + r) * H_ + h * HD_ + c));
        }
#pragma unroll
        for (int rep = 0; rep < 2; ++rep) {
            int o = tid + rep * 256;
            int r = o >> 3, c = (o & 7) * 8;        // r = d row, c = token offset
            async_ld_b128(voff + buf * 8192 + o * 16,
                          Vt + vbase + (size_t)r * S_ + s0 + c);
        }
        if (tid < 64) ML[buf][tid] = mask[b * S_ + s0 + tid];
    };

    // Prologue: Q tile (4 ops/wave) + chunk 0.
#pragma unroll
    for (int rep = 0; rep < 4; ++rep) {
        int o = tid + rep * 256;
        int r = o >> 3, c = (o & 7) * 8;
        async_ld_b128(qoff + o * 16,
                      Qb + ((size_t)(b * KSEL + qbase + r) * H_ + h * HD_ + c));
    }
    issue_chunk(0, 0);

    float mrun = -1.0e30f;
    float lrun = 0.0f;
    v8f acc2[4];
#pragma unroll
    for (int f = 0; f < 4; ++f) acc2[f] = zero8();

    const int NCH = S_ / 64;
    for (int it = 0; it < NCH; ++it) {
        const int buf = it & 1;
        if (it + 1 < NCH) { issue_chunk(buf ^ 1, (it + 1) * 64); S_WAIT_ASYNC_4; }
        else              { S_WAIT_ASYNC_0; }
        __syncthreads();

        const bf16_t* KLb = KL + buf * 4096;
        const bf16_t* VTb = VT + buf * 4096;

        // ---- scores^T: 4 key sub-tiles x (2 WMMA over d) ----
        float sv[4][8];
#pragma unroll
        for (int g = 0; g < 4; ++g) {
            v8f acc = zero8();
#pragma unroll
            for (int ds = 0; ds < HD_; ds += 32) {
                const bf16_t* ap = KLb + (g * 16 + col) * 64 + ds + hl * 8;
                v16bf A = ld_frag(ap, ap + 16);
                const bf16_t* bp = QL + (wave * 16 + col) * 64 + ds + hl * 8;
                v16bf Bq = ld_frag(bp, bp + 16);
                acc = __builtin_amdgcn_wmma_f32_16x16x32_bf16(
                    false, A, false, Bq, (short)0, acc, false, false);
            }
#pragma unroll
            for (int j = 0; j < 8; ++j)
                sv[g][j] = acc[j] * 0.125f + ML[buf][g * 16 + hl * 8 + j];
        }

        // ---- online softmax (keys split lane <-> lane^16) ----
        float mloc = -1.0e30f;
#pragma unroll
        for (int g = 0; g < 4; ++g)
#pragma unroll
            for (int j = 0; j < 8; ++j) mloc = fmaxf(mloc, sv[g][j]);
        mloc = fmaxf(mloc, swap16(mloc));
        float mnew = fmaxf(mrun, mloc);
        float corr = __expf(mrun - mnew);
        mrun = mnew;
        lrun *= corr;
#pragma unroll
        for (int f = 0; f < 4; ++f)
#pragma unroll
            for (int j = 0; j < 8; ++j) acc2[f][j] *= corr;

        float pv[4][8], pp[4][8];
#pragma unroll
        for (int g = 0; g < 4; ++g)
#pragma unroll
            for (int j = 0; j < 8; ++j) {
                float p = __expf(sv[g][j] - mnew);
                pv[g][j] = p;
                lrun += p;
                pp[g][j] = swap16(p);
            }

        // ---- ctx^T += V^T x P^T : 2 key-steps x 4 d-frags ----
#pragma unroll
        for (int s = 0; s < 64; s += 32) {
            const int g0 = s >> 4;
            v16bf P;
#pragma unroll
            for (int j = 0; j < 8; ++j) {
                float lo = hl ? pp[g0 + 1][j] : pv[g0][j];
                float hi = hl ? pv[g0 + 1][j] : pp[g0][j];
                P[j]     = (bf16_t)lo;
                P[j + 8] = (bf16_t)hi;
            }
#pragma unroll
            for (int f = 0; f < 4; ++f) {
                const bf16_t* ap = VTb + (f * 16 + col) * 64 + s + hl * 16;
                v16bf A2 = ld_frag(ap, ap + 8);
                acc2[f] = __builtin_amdgcn_wmma_f32_16x16x32_bf16(
                    false, A2, false, P, (short)0, acc2[f], false, false);
            }
        }
        __syncthreads();   // protect buffer reuse
    }

    float ltot = lrun + swap16(lrun);
    float inv  = (ltot > 0.0f) ? (1.0f / ltot) : 0.0f;
    const int qrow = b * KSEL + qbase + wave * 16 + col;
#pragma unroll
    for (int f = 0; f < 4; ++f) {
#pragma unroll
        for (int j = 0; j < 8; ++j) {
            int d = f * 16 + 8 * hl + j;
            out[(size_t)qrow * H_ + h * HD_ + d] = acc2[f][j] * inv;
        }
    }
}

// ---------------------------------------------------------------------------
// Host launcher.
// Inputs: 0 hidden 1 mask 2 remain 3 Wq 4 bq 5 Wk 6 bk 7 Wv 8 bv
// Workspace: idx/mul + bf16 hidden/weights + bf16 Q/K/Vt  (~62 MB).
// ---------------------------------------------------------------------------
extern "C" void kernel_launch(void* const* d_in, const int* in_sizes, int n_in,
                              void* d_out, int out_size, void* d_ws, size_t ws_size,
                              hipStream_t stream) {
    const float* hidden = (const float*)d_in[0];
    const float* amask  = (const float*)d_in[1];
    const int*   remain = (const int*)d_in[2];
    const float* Wq = (const float*)d_in[3];
    const float* bq = (const float*)d_in[4];
    const float* Wk = (const float*)d_in[5];
    const float* bk = (const float*)d_in[6];
    const float* Wv = (const float*)d_in[7];
    const float* bv = (const float*)d_in[8];
    float* out = (float*)d_out;
    (void)in_sizes; (void)n_in; (void)out_size; (void)ws_size;

    const size_t nH = (size_t)B_ * S_ * H_;   // 8M
    const size_t nW = (size_t)H_ * H_;        // 1M

    char* ws = (char*)d_ws;
    int*    gidx = (int*)ws;
    float*  gmul = (float*)(ws + 16384);
    bf16_t* Hbf  = (bf16_t*)(ws + 32768);
    bf16_t* Wqb  = Hbf + nH;
    bf16_t* Wkb  = Wqb + nW;
    bf16_t* Wvb  = Wkb + nW;
    bf16_t* Qb   = Wvb + nW;                       // B*K*H
    bf16_t* Kb   = Qb + (size_t)B_ * KSEL * H_;    // B*S*H
    bf16_t* Vtb  = Kb + nH;                        // (b,h,d,s)

    // 0) f32 -> bf16 pre-conversion
    cvt_bf16_kernel<<<(int)(nH / 4 / 256), 256, 0, stream>>>(hidden, Hbf, (int)nH);
    cvt_bf16_kernel<<<(int)(nW / 4 / 256), 256, 0, stream>>>(Wq, Wqb, (int)nW);
    cvt_bf16_kernel<<<(int)(nW / 4 / 256), 256, 0, stream>>>(Wk, Wkb, (int)nW);
    cvt_bf16_kernel<<<(int)(nW / 4 / 256), 256, 0, stream>>>(Wv, Wvb, (int)nW);

    // 1) token selection
    build_idx_kernel<<<B_, 32, 0, stream>>>(remain, gidx, gmul);

    // 2) projections
    gemm_bias_kernel<<<dim3(H_ / 64, (B_ * S_) / 128), 256, 0, stream>>>(
        Hbf, Wkb, bk, Kb, nullptr, nullptr, 0, 0);
    gemm_bias_kernel<<<dim3(H_ / 64, (B_ * S_) / 128), 256, 0, stream>>>(
        Hbf, Wvb, bv, Vtb, nullptr, nullptr, 0, 1);
    gemm_bias_kernel<<<dim3(H_ / 64, (B_ * KSEL) / 128), 256, 0, stream>>>(
        Hbf, Wqb, bq, Qb, gidx, gmul, KSEL, 0);

    // 3) fused flash attention
    attn_kernel<<<dim3(KSEL / 128, NH_, B_), 256, 0, stream>>>(
        Qb, Kb, Vtb, amask, out);
}
